// AdaptiveVectorQuantizer_86715389706646
// MI455X (gfx1250) — matrix operand order repaired and checked by reading, not compiled, not run on hardware
//
#include <hip/hip_runtime.h>
#include <hip/hip_bf16.h>

typedef __attribute__((ext_vector_type(16))) _Float16 v16h;
typedef __attribute__((ext_vector_type(8)))  float    v8f;

#define D        64
#define P        256
#define HWPIX    4096          // H*W
#define NPIX     (32 * HWPIX)  // B*H*W = 131072

__device__ __forceinline__ void cmin(float& bv, int& bi, float v, int i) {
    // argmin with lowest-index tie-break (matches jnp.argmin)
    if (v < bv || (v == bv && i < bi)) { bv = v; bi = i; }
}

// ---------------------------------------------------------------------------
// Kernel 1: scores = csq[j] - 2 * (x . c_j) via v_wmma_f32_16x16x32_f16,
// then all 8 prefix argmins in registers. One wave handles 16 pixels.
//   A (M axis)  = codebook entries (from LDS, f16)
//   B (N axis)  = pixels           (from global NCHW f32 -> f16)
//   C/D lane n  = all scores of pixel n  (entries ct*16 + hi*8 + r)
// ---------------------------------------------------------------------------
__global__ __launch_bounds__(256)
void avq_wmma_argmin_kernel(const float* __restrict__ x,
                            const float* __restrict__ cb,
                            unsigned long long* __restrict__ idx_out)
{
    __shared__ _Float16 cbh[P * D];   // codebook, f16, row-major
    __shared__ float    csq[P];       // per-entry squared norms, f32

    const int t = threadIdx.x;

    // Stage codebook: thread t owns entry t (64 floats -> f16 + squared norm).
    {
        const float* src = cb + t * D;
        float sq = 0.0f;
        #pragma unroll
        for (int c = 0; c < D; ++c) {
            float v = src[c];
            sq += v * v;
            cbh[t * D + c] = (_Float16)v;
        }
        csq[t] = sq;
    }
    __syncthreads();

    const int wave = t >> 5;
    const int lane = t & 31;
    const int lo   = lane & 15;      // position within half-wave
    const int hi   = lane >> 4;      // which half

    const int tile = blockIdx.x * 8 + wave;   // 16 pixels per wave
    const int row  = tile * 16 + lo;          // flat pixel index this lane owns (as B column)
    const int b    = row >> 12;
    const int hw   = row & (HWPIX - 1);
    const float* xbase = x + (size_t)b * D * HWPIX + hw;   // + c*HWPIX walks channels

    v8f acc[16] = {};   // 16 column tiles of 16 entries: full 256-entry score row

    #pragma unroll
    for (int ks = 0; ks < 2; ++ks) {          // K = 64 in two 32-steps
        const int K0 = ks * 32;

        // B fragment (pixels): lane holds K = K0 + hi*16 + (0..15), strided NCHW reads
        // (coalesced across the 16 lanes of each half for every element e).
        v16h bfrag;
        const int cb0 = K0 + hi * 16;
        #pragma unroll
        for (int e = 0; e < 16; ++e)
            bfrag[e] = (_Float16)xbase[(size_t)(cb0 + e) * HWPIX];

        #pragma unroll
        for (int ct = 0; ct < 16; ++ct) {
            // A fragment (codebook) from LDS: entry = ct*16 + lo,
            // elems 0..15 -> K0 + hi*16 + e.
            const int entry = ct * 16 + lo;
            const int k1    = K0 + hi * 16;
            v16h afrag;
            #pragma unroll
            for (int e = 0; e < 8; ++e) afrag[e]     = cbh[entry * D + k1 + e];
            #pragma unroll
            for (int e = 0; e < 8; ++e) afrag[8 + e] = cbh[entry * D + k1 + 8 + e];

            acc[ct] = __builtin_amdgcn_wmma_f32_16x16x32_f16(
                /*neg_a=*/false, afrag, /*neg_b=*/false, bfrag,
                /*c_mod=*/(short)0, acc[ct], /*reuse_a=*/false, /*reuse_b=*/false);
        }
    }

    // ---- prefix argmins over entries 0..255 (x_sq dropped: constant per pixel) ----
    int lvl_idx[8];
    float bv = 3.0e38f; int bi = 1 << 30;     // running min over entries < 16*(ct+1)

    #pragma unroll
    for (int ct = 0; ct < 16; ++ct) {
        float lv = 3.0e38f; int li = 1 << 30; // this lane's 8 entries of the block
        #pragma unroll
        for (int r = 0; r < 8; ++r) {
            const int entry = ct * 16 + hi * 8 + r;
            const float s   = csq[entry] - 2.0f * acc[ct][r];
            cmin(lv, li, s, entry);
            if (ct == 0) {                    // low levels: entries 0..1/3/7 (hi==0 lanes)
                if (r == 1) lvl_idx[0] = li;  // k = 2
                if (r == 3) lvl_idx[1] = li;  // k = 4
                if (r == 7) lvl_idx[2] = li;  // k = 8
            }
        }
        // combine the two halves of this 16-entry block
        const float ov = __shfl_xor(lv, 16, 32);
        const int   oi = __shfl_xor(li, 16, 32);
        cmin(lv, li, ov, oi);
        cmin(bv, bi, lv, li);                 // fold block into running prefix
        if (ct == 0)  lvl_idx[3] = bi;        // k = 16
        if (ct == 1)  lvl_idx[4] = bi;        // k = 32
        if (ct == 3)  lvl_idx[5] = bi;        // k = 64
        if (ct == 7)  lvl_idx[6] = bi;        // k = 128
        if (ct == 15) lvl_idx[7] = bi;        // k = 256
    }

    if (hi == 0) {                            // lane `lo` writes pixel `row`
        unsigned long long packed = 0;
        #pragma unroll
        for (int l = 0; l < 8; ++l)
            packed |= (unsigned long long)(lvl_idx[l] & 0xFF) << (8 * l);
        idx_out[row] = packed;
    }
}

// ---------------------------------------------------------------------------
// Kernel 2: bandwidth phase. out[lvl][b][c][hw] = cb[idx(lvl, b, hw)][c].
// Codebook staged TRANSPOSED in LDS so gather bank = idx & 63 (conflict-free
// for random indices). Writes are contiguous 1 KB runs per channel.
// Block = (lvl, b, 1024-pixel chunk); 256 threads, 4 pixels each.
// ---------------------------------------------------------------------------
__global__ __launch_bounds__(256)
void avq_scatter_kernel(const float* __restrict__ cb,
                        const unsigned long long* __restrict__ idxp,
                        float* __restrict__ out)
{
    __shared__ float cbT[D * P];   // cbT[c*256 + entry]

    const int t = threadIdx.x;

    // Stage transposed codebook: thread t owns entry t.
    {
        const float4* rowv = (const float4*)(cb + t * D);
        #pragma unroll
        for (int c4 = 0; c4 < 16; ++c4) {
            float4 v = rowv[c4];
            cbT[(c4 * 4 + 0) * P + t] = v.x;
            cbT[(c4 * 4 + 1) * P + t] = v.y;
            cbT[(c4 * 4 + 2) * P + t] = v.z;
            cbT[(c4 * 4 + 3) * P + t] = v.w;
        }
    }
    __syncthreads();

    const int bx    = blockIdx.x;
    const int chunk = bx & 3;              // 4 chunks of 1024 pixels
    const int b     = (bx >> 2) & 31;      // batch
    const int lvl   = bx >> 7;             // level 0..7
    const int pix0  = chunk * 1024;

    __builtin_prefetch(&idxp[b * HWPIX + pix0], 0, 0);

    int ids[4];
    #pragma unroll
    for (int p = 0; p < 4; ++p) {
        const int pix = pix0 + p * 256 + t;
        const unsigned long long pk = idxp[b * HWPIX + pix];
        ids[p] = (int)((pk >> (8 * lvl)) & 0xFF);
    }

    float* obase = out + (((size_t)lvl * 32 + b) * D) * HWPIX + pix0;
    #pragma unroll 4
    for (int c = 0; c < D; ++c) {
        float* oc = obase + (size_t)c * HWPIX;
        #pragma unroll
        for (int p = 0; p < 4; ++p)
            oc[p * 256 + t] = cbT[c * P + ids[p]];
    }
}

// ---------------------------------------------------------------------------
extern "C" void kernel_launch(void* const* d_in, const int* in_sizes, int n_in,
                              void* d_out, int out_size, void* d_ws, size_t ws_size,
                              hipStream_t stream)
{
    const float* x  = (const float*)d_in[0];   // (32,64,64,64) f32
    const float* cb = (const float*)d_in[1];   // (256,64) f32
    // d_in[2] previous_active_vectors: unused by the reference computation
    // d_in[3] num_active_vectors == 256 -> 8 levels, baked in

    unsigned long long* idxbuf = (unsigned long long*)d_ws;   // 131072 * 8 B = 1 MB

    // 1024 blocks x 8 waves x 16 pixels = 131072 pixels
    avq_wmma_argmin_kernel<<<1024, 256, 0, stream>>>(x, cb, idxbuf);

    // 8 lvl x 32 b x 4 chunks = 1024 blocks
    avq_scatter_kernel<<<1024, 256, 0, stream>>>(cb, idxbuf, (float*)d_out);
}
